// I2HOFI_79654463472299
// MI455X (gfx1250) — matrix operand order repaired and checked by reading, not compiled, not run on hardware
//
#include <hip/hip_runtime.h>
#include <hip/hip_bf16.h>
#include <math.h>

// ---------------------------------------------------------------------------
// I2HOFI GNN inference pipeline for MI455X (gfx1250), wave32 + WMMA f16.
// Weights are repacked once into native WMMA B-fragment layout (f16), so GEMM
// inner loops do only coalesced b128 loads + v_wmma. Graph-local stages use
// LDS tiles filled with global_load_async_to_lds_b128 (ASYNCcnt).
// ---------------------------------------------------------------------------

typedef __attribute__((ext_vector_type(16))) _Float16 v16h;
typedef __attribute__((ext_vector_type(8)))  float    v8f;

#define B_   32
#define G_   27
#define N1_  72
#define D_   256
#define C_   200
#define ROWS_     (B_ * G_ * N1_)   // 62208 feature rows per branch
#define NPB_      (2 * G_ * N1_)    // 3888 pooled nodes per batch element
#define POOLROWS_ (B_ * NPB_)       // 124416

__device__ __forceinline__ float sigmoid_(float x) { return 1.0f / (1.0f + __expf(-x)); }

// ---- CDNA5 async global->LDS copy (per-lane 16B), tracked by ASYNCcnt ------
__device__ __forceinline__ void async_ld_b128(void* lds_dst, const void* gsrc) {
    // LDS aperture: low 32 bits of the flat shared address are the LDS offset.
    unsigned loff = (unsigned)(unsigned long long)(uintptr_t)lds_dst;
    unsigned long long ga = (unsigned long long)(uintptr_t)gsrc;
    asm volatile("global_load_async_to_lds_b128 %0, %1, off"
                 :: "v"(loff), "v"(ga) : "memory");
}
__device__ __forceinline__ void async_wait0() {
    asm volatile("s_wait_asynccnt 0" ::: "memory");
}

// ---------------------------------------------------------------------------
// Repack W[K,N] (f32, row-major) into f16 WMMA B fragments.
// Fragment (tn, kb): lane l, element e -> W[kb*32 + (l>=16)*8 + (e&7) + (e>=8)*16, tn*16 + (l&15)]
// Stored at Wp[((tn*(K/32) + kb)*32 + l)*16 + e]  (32B contiguous per lane).
// ---------------------------------------------------------------------------
__global__ void k_packW(const float* __restrict__ W, _Float16* __restrict__ Wp,
                        int K, int N) {
    const int t = blockIdx.x * blockDim.x + threadIdx.x;
    const int lane = t & 31;
    const int frag = t >> 5;
    const int ksteps = K >> 5;
    if (frag >= (N >> 4) * ksteps) return;
    const int tn = frag / ksteps;
    const int kb = frag - tn * ksteps;
    const int col = tn * 16 + (lane & 15);
    const int kbase = (kb << 5) + ((lane >> 4) << 3);
    _Float16* o = Wp + ((size_t)frag * 32 + lane) * 16;
#pragma unroll
    for (int e = 0; e < 16; ++e)
        o[e] = (_Float16)W[(size_t)(kbase + (e & 7) + ((e >> 3) << 4)) * N + col];
}

// ---------------------------------------------------------------------------
// GEMM: C[M,N] = act(A[M,K] @ W[K,N] + bias). 16x64 per wave (4 accumulators
// reuse one A fragment). Wp is the packed-f16 weight. ACT: 0 none, 1 sigmoid.
// ---------------------------------------------------------------------------
template <int ACT>
__global__ void k_gemm_wmma(const float* __restrict__ A, const _Float16* __restrict__ Wp,
                            const float* __restrict__ bias, float* __restrict__ C,
                            int M, int N, int K, int totalTiles) {
    const int lane = threadIdx.x & 31;
    const int wave = threadIdx.x >> 5;
    const int tile = blockIdx.x * (blockDim.x >> 5) + wave;
    if (tile >= totalTiles) return;                 // wave-uniform guard

    const int ngroups = N >> 6;                     // 64-column groups
    const int tm = tile / ngroups;
    const int tg = tile - tm * ngroups;
    const int ksteps = K >> 5;
    const int l15 = lane & 15;
    const int kbA = (lane >> 4) << 3;

    const float* Arow = A + (size_t)(tm * 16 + l15) * K + kbA;
    const v16h* bp = (const v16h*)Wp;

    v8f acc[4] = {{}, {}, {}, {}};
    for (int kb = 0; kb < ksteps; ++kb) {
        const float* Ap = Arow + (kb << 5);
        __builtin_prefetch((const void*)(Ap + 32), 0, 3);
        v16h af;
#pragma unroll
        for (int j = 0; j < 8; ++j) {
            af[j]     = (_Float16)Ap[j];            // K = kbA + 0..7
            af[8 + j] = (_Float16)Ap[16 + j];       // K = kbA + 16..23
        }
        v16h bb[4];
#pragma unroll
        for (int s = 0; s < 4; ++s)
            bb[s] = bp[((size_t)(tg * 4 + s) * ksteps + kb) * 32 + lane];
#pragma unroll
        for (int s = 0; s < 4; ++s)
            acc[s] = __builtin_amdgcn_wmma_f32_16x16x32_f16(false, af, false, bb[s],
                                                            (short)0, acc[s], false, false);
    }

    const int rbase = tm * 16 + ((lane >> 4) << 3);
#pragma unroll
    for (int s = 0; s < 4; ++s) {
        const int col = (tg * 4 + s) * 16 + l15;
        const float bv = bias ? bias[col] : 0.0f;
#pragma unroll
        for (int r = 0; r < 8; ++r) {
            float v = acc[s][r] + bv;
            if (ACT == 1) v = sigmoid_(v);
            C[(size_t)(rbase + r) * N + col] = v;
        }
    }
}

// ---------------------------------------------------------------------------
// Transpose x1 [B,G,N1,D] -> xi [B,N1,G,D]
// ---------------------------------------------------------------------------
__global__ void k_transpose(const float* __restrict__ x1, float* __restrict__ xi) {
    size_t i = (size_t)blockIdx.x * blockDim.x + threadIdx.x;
    int d = (int)(i & (D_ - 1));
    size_t t = i >> 8;
    int g = (int)(t % G_); t /= G_;
    int n = (int)(t % N1_);
    int b = (int)(t / N1_);
    xi[i] = x1[(((size_t)(b * G_ + g) * N1_ + n) << 8) + d];
}

// ---------------------------------------------------------------------------
// APPNP combine: y = 0.9 * (a @ h) + 0.1 * h + x  (graphs of N nodes, D=256).
// One block (256 threads) per graph; h tile streamed to LDS asynchronously.
// ---------------------------------------------------------------------------
template <int N>
__global__ void k_appnp(const float* __restrict__ h, const float* __restrict__ x,
                        const float* __restrict__ a, float* __restrict__ y) {
    __shared__ __align__(16) float sh[N * D_];
    __shared__ float sa[N * N];
    const int g = blockIdx.x;
    const float* hg = h + (size_t)g * N * D_;
    for (int i = threadIdx.x; i < N * (D_ / 4); i += blockDim.x)
        async_ld_b128(&sh[i * 4], hg + i * 4);
    for (int i = threadIdx.x; i < N * N; i += blockDim.x) sa[i] = a[i];
    async_wait0();
    __syncthreads();
    const int d = threadIdx.x;   // blockDim.x == 256 == D_
    for (int n = 0; n < N; ++n) {
        float s = 0.0f;
        for (int m = 0; m < N; ++m) s += sa[n * N + m] * sh[m * D_ + d];
        size_t o = (size_t)g * N * D_ + (size_t)n * D_ + d;
        y[o] = 0.9f * s + 0.1f * sh[n * D_ + d] + x[o];
    }
}

// ---------------------------------------------------------------------------
// Dense GAT (H=1): attention softmax + aggregate + bias + elu + residual.
// One block (256 threads) per graph. Output goes into concatenated x2 at
//   row = b*3888 + half_off + (g % gpb)*N + n.
// ---------------------------------------------------------------------------
template <int N>
__global__ void k_gat(const float* __restrict__ xin, const float* __restrict__ xp,
                      const float* __restrict__ a, const float* __restrict__ aks,
                      const float* __restrict__ akn, const float* __restrict__ bias,
                      float* __restrict__ out, int gpb, int half_off) {
    __shared__ __align__(16) float sxp[N * D_];
    __shared__ float satt[N * N];
    __shared__ float ss[N];
    __shared__ float st[N];
    const int g = blockIdx.x;
    const float* xpg = xp + (size_t)g * N * D_;
    for (int i = threadIdx.x; i < N * (D_ / 4); i += blockDim.x)
        async_ld_b128(&sxp[i * 4], xpg + i * 4);
    async_wait0();
    __syncthreads();

    // per-node attention features
    for (int n = threadIdx.x; n < N; n += blockDim.x) {
        float s = 0.0f, t = 0.0f;
        for (int d = 0; d < D_; ++d) {
            float v = sxp[n * D_ + d];
            s += v * aks[d];
            t += v * akn[d];
        }
        ss[n] = s; st[n] = t;
    }
    __syncthreads();

    // softmax rows of leaky_relu(ss[n]+st[m], 0.2) + mask
    for (int n = threadIdx.x; n < N; n += blockDim.x) {
        float mx = -1e30f;
        for (int m = 0; m < N; ++m) {
            float av = (m == n) ? 1.0f : a[n * N + m];   // add_self_loops
            float c  = ss[n] + st[m];
            c = (c > 0.0f) ? c : 0.2f * c;
            if (av == 0.0f) c += -1e10f;
            satt[n * N + m] = c;
            mx = fmaxf(mx, c);
        }
        float sum = 0.0f;
        for (int m = 0; m < N; ++m) {
            float e = __expf(satt[n * N + m] - mx);
            satt[n * N + m] = e;
            sum += e;
        }
        float inv = 1.0f / sum;
        for (int m = 0; m < N; ++m) satt[n * N + m] *= inv;
    }
    __syncthreads();

    const int d  = threadIdx.x;            // 256 == D_
    const int b  = g / gpb;
    const int gi = g - b * gpb;
    const float bv = bias[d];
    for (int n = 0; n < N; ++n) {
        float acc = 0.0f;
        for (int m = 0; m < N; ++m) acc += satt[n * N + m] * sxp[m * D_ + d];
        float v = acc + bv;
        v = (v > 0.0f) ? v : expm1f(v);    // elu
        size_t xo = (size_t)g * N * D_ + (size_t)n * D_ + d;
        v += xin[xo];
        size_t orow = (size_t)b * NPB_ + half_off + (size_t)gi * N + n;
        out[orow * D_ + d] = v;
    }
}

// ---------------------------------------------------------------------------
// Fused attention pool: xf[b,:] += sigmoid(X@Wg+bg) * (X@Wf+bf) over nodes.
// 16x64 tile per wave, feats+gate -> 8 WMMAs per K-step sharing the A fragment.
// X: [B*3888, 256], packed Wf/Wg: [256,512] fragments, xf: [B,512] (zeroed).
// ---------------------------------------------------------------------------
__global__ void k_pool(const float* __restrict__ X, const _Float16* __restrict__ WfP,
                       const float* __restrict__ bf, const _Float16* __restrict__ WgP,
                       const float* __restrict__ bg, float* __restrict__ xf,
                       int totalTiles) {
    const int lane = threadIdx.x & 31;
    const int wave = threadIdx.x >> 5;
    const int tile = blockIdx.x * (blockDim.x >> 5) + wave;
    if (tile >= totalTiles) return;

    const int tm = tile >> 3;            // row tile (16 rows; 243 tiles per batch)
    const int tg = tile & 7;             // 8 groups of 64 columns over N=512
    const int ksteps = D_ >> 5;          // 8
    const int l15 = lane & 15;
    const int kbA = (lane >> 4) << 3;

    const float* Arow = X + (size_t)(tm * 16 + l15) * D_ + kbA;
    const v16h* bfp = (const v16h*)WfP;
    const v16h* bgp = (const v16h*)WgP;

    v8f accF[4] = {{}, {}, {}, {}};
    v8f accG[4] = {{}, {}, {}, {}};
    for (int kb = 0; kb < ksteps; ++kb) {
        const float* Ap = Arow + (kb << 5);
        __builtin_prefetch((const void*)(Ap + 32), 0, 3);
        v16h af;
#pragma unroll
        for (int j = 0; j < 8; ++j) {
            af[j]     = (_Float16)Ap[j];
            af[8 + j] = (_Float16)Ap[16 + j];
        }
        v16h bF[4], bG[4];
#pragma unroll
        for (int s = 0; s < 4; ++s) {
            size_t idx = ((size_t)(tg * 4 + s) * ksteps + kb) * 32 + lane;
            bF[s] = bfp[idx];
            bG[s] = bgp[idx];
        }
#pragma unroll
        for (int s = 0; s < 4; ++s) {
            accF[s] = __builtin_amdgcn_wmma_f32_16x16x32_f16(false, af, false, bF[s],
                                                             (short)0, accF[s], false, false);
            accG[s] = __builtin_amdgcn_wmma_f32_16x16x32_f16(false, af, false, bG[s],
                                                             (short)0, accG[s], false, false);
        }
    }

    const int b = tm / (NPB_ / 16);      // 243 row tiles per batch
#pragma unroll
    for (int s = 0; s < 4; ++s) {
        const int col = (tg * 4 + s) * 16 + l15;
        const float fb = bf[col], gb = bg[col];
        float ssum = 0.0f;
#pragma unroll
        for (int r = 0; r < 8; ++r) {
            float fv = accF[s][r] + fb;
            float gv = sigmoid_(accG[s][r] + gb);
            ssum += fv * gv;
        }
        ssum += __shfl_xor(ssum, 16, 32);   // fold rows 8..15 onto 0..7 holders
        if (lane < 16) atomicAdd(&xf[b * (2 * D_) + col], ssum);
    }
}

__global__ void k_zero(float* __restrict__ p, int n) {
    int i = blockIdx.x * blockDim.x + threadIdx.x;
    if (i < n) p[i] = 0.0f;
}

// ---------------------------------------------------------------------------
// Head: BatchNorm(inference) + dense [512,200] + softmax. One block per batch.
// ---------------------------------------------------------------------------
__global__ void k_head(const float* __restrict__ xf, const float* __restrict__ gamma,
                       const float* __restrict__ beta, const float* __restrict__ mean,
                       const float* __restrict__ var, const float* __restrict__ Wd,
                       const float* __restrict__ bd, float* __restrict__ out) {
    __shared__ float sx[2 * D_];
    __shared__ float red[256];
    const int b = blockIdx.x, t = threadIdx.x;
    for (int i = t; i < 2 * D_; i += 256) {
        float v = xf[b * 2 * D_ + i];
        sx[i] = (v - mean[i]) * rsqrtf(var[i] + 1e-3f) * gamma[i] + beta[i];
    }
    __syncthreads();
    float lg = 0.0f;
    if (t < C_) {
        float s = bd[t];
        for (int k = 0; k < 2 * D_; ++k) s += sx[k] * Wd[k * C_ + t];
        lg = s;
    }
    red[t] = (t < C_) ? lg : -1e30f;
    __syncthreads();
    for (int o = 128; o > 0; o >>= 1) { if (t < o) red[t] = fmaxf(red[t], red[t + o]); __syncthreads(); }
    const float mx = red[0];
    __syncthreads();
    const float e = (t < C_) ? __expf(lg - mx) : 0.0f;
    red[t] = e;
    __syncthreads();
    for (int o = 128; o > 0; o >>= 1) { if (t < o) red[t] += red[t + o]; __syncthreads(); }
    const float inv = 1.0f / red[0];
    if (t < C_) out[b * C_ + t] = e * inv;
}

// ---------------------------------------------------------------------------
template <int ACT>
static void launch_gemm(const float* A, const _Float16* Wp, const float* bias,
                        float* C, int M, int N, int K, hipStream_t stream) {
    int tiles = (M / 16) * (N / 64);
    int blocks = (tiles + 3) / 4;
    k_gemm_wmma<ACT><<<blocks, 128, 0, stream>>>(A, Wp, bias, C, M, N, K, tiles);
}

static void launch_pack(const float* W, _Float16* Wp, int K, int N, hipStream_t stream) {
    int threads = (N / 16) * (K / 32) * 32;
    k_packW<<<(threads + 255) / 256, 256, 0, stream>>>(W, Wp, K, N);
}

extern "C" void kernel_launch(void* const* d_in, const int* in_sizes, int n_in,
                              void* d_out, int out_size, void* d_ws, size_t ws_size,
                              hipStream_t stream) {
    const float* x1      = (const float*)d_in[0];
    const float* a_intra = (const float*)d_in[1];
    const float* a_inter = (const float*)d_in[2];
    const float* gcn_w   = (const float*)d_in[3];
    const float* gcn_b   = (const float*)d_in[4];
    const float* gat_k   = (const float*)d_in[5];   // [256,1,256] == [256,256]
    const float* ak_s    = (const float*)d_in[6];
    const float* ak_n    = (const float*)d_in[7];
    const float* gat_b   = (const float*)d_in[8];
    const float* pf_w    = (const float*)d_in[9];
    const float* pf_b    = (const float*)d_in[10];
    const float* pa_w    = (const float*)d_in[11];
    const float* pa_b    = (const float*)d_in[12];
    const float* bn_g    = (const float*)d_in[13];
    const float* bn_be   = (const float*)d_in[14];
    const float* bn_m    = (const float*)d_in[15];
    const float* bn_v    = (const float*)d_in[16];
    const float* dw      = (const float*)d_in[17];
    const float* db      = (const float*)d_in[18];
    float* out = (float*)d_out;

    const size_t SZ = (size_t)ROWS_ * D_;            // 15.9M floats
    float* ws = (float*)d_ws;
    float* xi = ws;                                  // [B,N1,G,D]
    float* h  = ws + SZ;                             // APPNP h, reused as GAT xp
    float* y  = ws + 2 * SZ;                         // post-APPNP features
    float* x2 = ws + 3 * SZ;                         // [B,3888,256] (2*SZ floats)
    float* xf = ws + 5 * SZ;                         // [B,512]
    _Float16* pk = (_Float16*)(ws + 5 * SZ + B_ * 2 * D_);
    _Float16* pk_gcn = pk;                           // 16*8*32*16  = 65536 f16
    _Float16* pk_gat = pk + 65536;
    _Float16* pk_pf  = pk + 2 * 65536;               // 32*8*32*16 = 131072 f16
    _Float16* pk_pa  = pk + 2 * 65536 + 131072;

    // one-shot weight repack into WMMA fragment layout
    launch_pack(gcn_w, pk_gcn, D_, D_, stream);
    launch_pack(gat_k, pk_gat, D_, D_, stream);
    launch_pack(pf_w,  pk_pf,  D_, 2 * D_, stream);
    launch_pack(pa_w,  pk_pa,  D_, 2 * D_, stream);

    // x1 -> xi transpose
    k_transpose<<<(unsigned)(SZ / 256), 256, 0, stream>>>(x1, xi);

    // ---- intra branch (graphs of N1=72 over ROI axis) ----
    launch_gemm<1>(x1, pk_gcn, gcn_b, h, ROWS_, D_, D_, stream);
    k_appnp<N1_><<<B_ * G_, 256, 0, stream>>>(h, x1, a_intra, y);
    launch_gemm<0>(y, pk_gat, nullptr, h, ROWS_, D_, D_, stream);
    k_gat<N1_><<<B_ * G_, 256, 0, stream>>>(y, h, a_intra, ak_s, ak_n, gat_b, x2, G_, 0);

    // ---- inter branch (graphs of G=27 over channel-node axis) ----
    launch_gemm<1>(xi, pk_gcn, gcn_b, h, ROWS_, D_, D_, stream);
    k_appnp<G_><<<B_ * N1_, 256, 0, stream>>>(h, xi, a_inter, y);
    launch_gemm<0>(y, pk_gat, nullptr, h, ROWS_, D_, D_, stream);
    k_gat<G_><<<B_ * N1_, 256, 0, stream>>>(y, h, a_inter, ak_s, ak_n, gat_b, x2,
                                            N1_, G_ * N1_);

    // ---- fused attention pooling ----
    k_zero<<<(B_ * 2 * D_ + 255) / 256, 256, 0, stream>>>(xf, B_ * 2 * D_);
    {
        const int tiles = (POOLROWS_ / 16) * 8;      // 62208 wave tiles
        k_pool<<<(tiles + 3) / 4, 128, 0, stream>>>(x2, pk_pf, pf_b, pk_pa, pa_b,
                                                    xf, tiles);
    }

    // ---- BN + dense + softmax ----
    k_head<<<B_, 256, 0, stream>>>(xf, bn_g, bn_be, bn_m, bn_v, dw, db, out);
}
